// MCCDecoderBlock_10892037063170
// MI455X (gfx1250) — matrix operand order repaired and checked by reading, not compiled, not run on hardware
//
#include <hip/hip_runtime.h>
#include <hip/hip_fp16.h>

#define DIM 768
#define HEADS 12
#define HEAD_DIM 64
#define NPATCH 16
#define PTOK 2048
#define QTOK 1024
#define NTOK (NPATCH + PTOK + QTOK) /* 3088 */
#define HIDDEN (4 * DIM)            /* 3072 */
#define CTX (NTOK - QTOK)           /* 2064 */

typedef __attribute__((ext_vector_type(16))) _Float16 v16h;
typedef __attribute__((ext_vector_type(8)))  _Float16 v8h;
typedef __attribute__((ext_vector_type(8)))  float    v8f;
typedef __attribute__((ext_vector_type(4)))  int      v4i;

#if defined(__has_builtin)
#if __has_builtin(__builtin_amdgcn_global_load_async_to_lds_b128)
#define HAVE_ASYNC_LDS 1
#else
#define HAVE_ASYNC_LDS 0
#endif
#else
#define HAVE_ASYNC_LDS 0
#endif

// One 16-byte global->LDS move per lane. Async path uses the CDNA5
// GLOBAL_LOAD_ASYNC_TO_LDS_B128 (ASYNCcnt-tracked, no VGPR round trip).
__device__ __forceinline__ void copy_b128_g2l(const _Float16* g, _Float16* l) {
#if HAVE_ASYNC_LDS
  __builtin_amdgcn_global_load_async_to_lds_b128(
      (v4i*)const_cast<_Float16*>(g), (v4i*)l, 0, 0);
#else
  *reinterpret_cast<v8h*>(l) = *reinterpret_cast<const v8h*>(g);
#endif
}

__device__ __forceinline__ void wait_tiles_and_barrier() {
#if HAVE_ASYNC_LDS
#if __has_builtin(__builtin_amdgcn_s_wait_asynccnt)
  __builtin_amdgcn_s_wait_asynccnt(0);
#else
  asm volatile("s_wait_asynccnt 0" ::: "memory");
#endif
#endif
  __syncthreads();
}

__device__ __forceinline__ v8f wmma16x16x32(v16h a, v16h b, v8f c) {
  // D = A(16x32 f16) * B(32x16 f16) + C(16x16 f32)
  return __builtin_amdgcn_wmma_f32_16x16x32_f16(false, a, false, b, (short)0, c,
                                                false, false);
}

// A-fragment (16x32 f16): lane L: m=L&15, half=L>>4; elems 0..15 -> K=half*16+i
// (K-contiguous, matching the B-fragment).
__device__ __forceinline__ v16h lds_afrag(const _Float16* base, int stride,
                                          int row0, int lane) {
  const int m = lane & 15, h = (lane >> 4) & 1;
  const _Float16* p = base + (row0 + m) * stride + h * 16;
  v16h r;
#pragma unroll
  for (int i = 0; i < 16; ++i) r[i] = p[i];
  return r;
}

// B-fragment (32x16 f16): lane L: n=L&15, K = (L>>4)*16 + i (K-contiguous).
__device__ __forceinline__ v16h lds_bfrag(const _Float16* base, int stride,
                                          int col0, int lane) {
  const int n = lane & 15, klo = ((lane >> 4) & 1) * 16;
  const _Float16* p = base + (col0 + n) * stride + klo;
  v16h r;
#pragma unroll
  for (int i = 0; i < 16; ++i) r[i] = p[i];
  return r;
}

// ---------------------------------------------------------------- utilities
__global__ void f32_to_f16_kernel(const float* __restrict__ in,
                                  _Float16* __restrict__ out, int n) {
  int i = blockIdx.x * blockDim.x + threadIdx.x;
  if (i < n) out[i] = (_Float16)in[i];
}

__global__ void seg_minmax_kernel(const int* __restrict__ patch_idx,
                                  int* __restrict__ lohi) {
  __shared__ int smin[NPATCH], smax[NPATCH];
  if (threadIdx.x < NPATCH) { smin[threadIdx.x] = 0x7fffffff; smax[threadIdx.x] = -1; }
  __syncthreads();
  for (int i = threadIdx.x; i < PTOK; i += blockDim.x) {
    int o = patch_idx[i];
    atomicMin(&smin[o], i);
    atomicMax(&smax[o], i);
  }
  __syncthreads();
  if (threadIdx.x < NPATCH) {
    lohi[threadIdx.x] = smin[threadIdx.x] + NPATCH;          // pmin (col space)
    lohi[NPATCH + threadIdx.x] = smax[threadIdx.x] + NPATCH; // pmax (col space)
  }
}

__global__ __launch_bounds__(256) void layernorm_f16_kernel(
    const float* __restrict__ x, const float* __restrict__ w,
    const float* __restrict__ b, _Float16* __restrict__ out) {
  const int row = blockIdx.x, tid = threadIdx.x;
  const float* xr = x + (size_t)row * DIM;
  float v0 = xr[tid], v1 = xr[tid + 256], v2 = xr[tid + 512];
  __shared__ float red[256], red2[256];
  red[tid] = v0 + v1 + v2;
  red2[tid] = v0 * v0 + v1 * v1 + v2 * v2;
  __syncthreads();
  for (int off = 128; off > 0; off >>= 1) {
    if (tid < off) { red[tid] += red[tid + off]; red2[tid] += red2[tid + off]; }
    __syncthreads();
  }
  float mu = red[0] * (1.0f / DIM);
  float var = red2[0] * (1.0f / DIM) - mu * mu;
  float rs = rsqrtf(var + 1e-5f);
  _Float16* orow = out + (size_t)row * DIM;
  orow[tid]       = (_Float16)((v0 - mu) * rs * w[tid] + b[tid]);
  orow[tid + 256] = (_Float16)((v1 - mu) * rs * w[tid + 256] + b[tid + 256]);
  orow[tid + 512] = (_Float16)((v2 - mu) * rs * w[tid + 512] + b[tid + 512]);
}

// ---------------------------------------------------------------- WMMA GEMM
// C[M x Nc] = A[M x K] * B[Nc x K]^T , f16 in / f32 accumulate.
// 128x128 block tile, K-step 64, double-buffered LDS (one barrier per step);
// 8 waves, wave = 32 rows x 64 cols.
enum { EPI_QKV = 0, EPI_PROJ = 1, EPI_FC1 = 2, EPI_FC2 = 3 };

template <int EPI>
__global__ __launch_bounds__(256) void gemm_wmma_kernel(
    const _Float16* __restrict__ A, const _Float16* __restrict__ B, int M,
    int K, int Ncols, const float* __restrict__ bias,
    const float* __restrict__ resid, float* __restrict__ outF,
    _Float16* __restrict__ outH, _Float16* __restrict__ q16,
    _Float16* __restrict__ k16, _Float16* __restrict__ v16) {
  __shared__ __align__(16) _Float16 As[2][128 * 72]; // 64 halfs + 8 pad / row
  __shared__ __align__(16) _Float16 Bs[2][128 * 72];
  const int tid = threadIdx.x;
  const int wid = tid >> 5, lane = tid & 31;
  const int rowBlk = blockIdx.x * 128, colBlk = blockIdx.y * 128;
  const int wm = (wid & 3) * 32, wn = (wid >> 2) * 64;

  const v8f vzero = {0.f, 0.f, 0.f, 0.f, 0.f, 0.f, 0.f, 0.f};
  v8f acc[2][4];
#pragma unroll
  for (int i = 0; i < 2; ++i)
#pragma unroll
    for (int j = 0; j < 4; ++j) acc[i][j] = vzero;

  const int ldr = tid >> 1;        // 0..127 : LDS tile row
  const int seg = (tid & 1) * 32;  // half offset within the 64-wide K slab
  // clamp rather than zero-pad: rows >= M produce outputs that are never stored
  int gra = rowBlk + ldr; if (gra >= M) gra = M - 1;
  const int grb = colBlk + ldr; // N dims are multiples of 128 -> always valid
  const _Float16* gA = A + (size_t)gra * K + seg;
  const _Float16* gB = B + (size_t)grb * K + seg;

  auto issue_tile = [&](int kb, int buf) {
#pragma unroll
    for (int i = 0; i < 4; ++i)
      copy_b128_g2l(gA + kb + i * 8, &As[buf][ldr * 72 + seg + i * 8]);
#pragma unroll
    for (int i = 0; i < 4; ++i)
      copy_b128_g2l(gB + kb + i * 8, &Bs[buf][ldr * 72 + seg + i * 8]);
  };

  const int nk = K >> 6;
  issue_tile(0, 0);
  for (int it = 0; it < nk; ++it) {
    const int p = it & 1;
    wait_tiles_and_barrier(); // tile `it` resident; prev reads of buf p^1 done
    if (it + 1 < nk) issue_tile((it + 1) << 6, p ^ 1);
    const _Float16* as = As[p];
    const _Float16* bs = Bs[p];
#pragma unroll
    for (int ks = 0; ks < 2; ++ks) {
      v16h a0 = lds_afrag(as + ks * 32, 72, wm, lane);
      v16h a1 = lds_afrag(as + ks * 32, 72, wm + 16, lane);
#pragma unroll
      for (int j = 0; j < 4; ++j) {
        v16h b = lds_bfrag(bs + ks * 32, 72, wn + j * 16, lane);
        acc[0][j] = wmma16x16x32(a0, b, acc[0][j]);
        acc[1][j] = wmma16x16x32(a1, b, acc[1][j]);
      }
    }
  }

  const int hl = lane >> 4, ln = lane & 15;
#pragma unroll
  for (int i = 0; i < 2; ++i)
#pragma unroll
    for (int j = 0; j < 4; ++j)
#pragma unroll
      for (int r = 0; r < 8; ++r) {
        int row = rowBlk + wm + i * 16 + r + 8 * hl;
        int col = colBlk + wn + j * 16 + ln;
        if (row >= M) continue;
        float v = acc[i][j][r];
        if (EPI == EPI_QKV) {
          int which = col / DIM, rem = col % DIM;
          int h = rem / HEAD_DIM, d = rem % HEAD_DIM;
          _Float16* dst = (which == 0) ? q16 : (which == 1) ? k16 : v16;
          dst[((size_t)h * NTOK + row) * HEAD_DIM + d] = (_Float16)v;
        } else if (EPI == EPI_PROJ) {
          outF[(size_t)row * Ncols + col] =
              v + bias[col] + resid[(size_t)row * Ncols + col];
        } else if (EPI == EPI_FC1) {
          float t = v + bias[col];
          float g = 0.5f * t * (1.0f + erff(t * 0.70710678118654752f));
          outH[(size_t)row * Ncols + col] = (_Float16)g;
        } else { // EPI_FC2
          outF[(size_t)row * Ncols + col] =
              v + bias[col] + resid[(size_t)row * Ncols + col];
        }
      }
}

// ------------------------------------------------------- flash attention
// grid (rowBlocks=25, heads=12), 256 threads = 8 waves, wave owns 16 rows.
__global__ __launch_bounds__(256) void attn_flash_kernel(
    const _Float16* __restrict__ q16, const _Float16* __restrict__ k16,
    const _Float16* __restrict__ v16, const int* __restrict__ point_idx,
    const int* __restrict__ lohi, _Float16* __restrict__ attn16) {
  __shared__ __align__(16) _Float16 Qs[128 * 72];
  __shared__ __align__(16) _Float16 Ks[64 * 72];
  __shared__ __align__(16) _Float16 Vts[64 * 72]; // transposed: [d][kv]
  __shared__ __align__(16) _Float16 Ps[8 * 16 * 72];
  __shared__ int rp_lo[128], rp_hi[128], rp_e1[128], rp_e2[128];

  const int tid = threadIdx.x;
  const int wid = tid >> 5, lane = tid & 31;
  const int hl = lane >> 4, ln = lane & 15;
  const int head = blockIdx.y;
  const int rowBlk = blockIdx.x * 128;
  const _Float16* qh = q16 + (size_t)head * NTOK * HEAD_DIM;
  const _Float16* kh = k16 + (size_t)head * NTOK * HEAD_DIM;
  const _Float16* vh = v16 + (size_t)head * NTOK * HEAD_DIM;

  { // load Q tile (128 x 64); clamped rows: garbage rows are fully masked
    int r = tid >> 1, seg = (tid & 1) * 32;
    int gr = rowBlk + r; if (gr >= NTOK) gr = NTOK - 1;
    const v8h* p =
        reinterpret_cast<const v8h*>(qh + (size_t)gr * HEAD_DIM + seg);
#pragma unroll
    for (int i = 0; i < 4; ++i)
      *reinterpret_cast<v8h*>(&Qs[r * 72 + seg + i * 8]) = p[i];
  }
  if (tid < 128) { // per-row mask params: allowed = [lo,hi] || ==e1 || ==e2
    int row = rowBlk + tid;
    int lo, hi, e1, e2;
    if (row < CTX) { lo = 0; hi = CTX - 1; e1 = -1; e2 = -1; }
    else if (row < NTOK) {
      int o = point_idx[row - CTX];
      lo = lohi[o]; hi = lohi[NPATCH + o]; e1 = row; e2 = o;
    } else { lo = 1; hi = 0; e1 = -1; e2 = -1; } // padded row: all masked
    rp_lo[tid] = lo; rp_hi[tid] = hi; rp_e1[tid] = e1; rp_e2[tid] = e2;
  }

  const int waveRow = wid * 16;
  const v8f vzero = {0.f, 0.f, 0.f, 0.f, 0.f, 0.f, 0.f, 0.f};
  float m_run[8], l_run[8];
  v8f o_acc[4];
#pragma unroll
  for (int r = 0; r < 8; ++r) { m_run[r] = -1e30f; l_run[r] = 0.0f; }
#pragma unroll
  for (int j = 0; j < 4; ++j) o_acc[j] = vzero;

  for (int cb = 0; cb < NTOK; cb += 64) {
    __syncthreads();
    if (tid < 128) { // K tile (64 x 64); clamped (masked cols kill garbage)
      int r = tid >> 1, seg = (tid & 1) * 32;
      int gr = cb + r; if (gr >= NTOK) gr = NTOK - 1;
      const v8h* p =
          reinterpret_cast<const v8h*>(kh + (size_t)gr * HEAD_DIM + seg);
#pragma unroll
      for (int i = 0; i < 4; ++i)
        *reinterpret_cast<v8h*>(&Ks[r * 72 + seg + i * 8]) = p[i];
    }
    { // V tile transposed into Vts[d][kv]; clamped (P=0 kills garbage)
      int i = tid & 63, dg = (tid >> 6) * 16;
      int gr = cb + i; if (gr >= NTOK) gr = NTOK - 1;
      const v8h* p =
          reinterpret_cast<const v8h*>(vh + (size_t)gr * HEAD_DIM + dg);
      v8h va = p[0], vb = p[1];
#pragma unroll
      for (int j = 0; j < 8; ++j) Vts[(dg + j) * 72 + i] = va[j];
#pragma unroll
      for (int j = 0; j < 8; ++j) Vts[(dg + 8 + j) * 72 + i] = vb[j];
    }
    __syncthreads();

    // S(16x64) = Q K^T
    v16h qa0 = lds_afrag(Qs, 72, waveRow, lane);
    v16h qa1 = lds_afrag(Qs + 32, 72, waveRow, lane);
    v8f s[4];
#pragma unroll
    for (int t = 0; t < 4; ++t) {
      v16h b0 = lds_bfrag(Ks, 72, t * 16, lane);
      v16h b1 = lds_bfrag(Ks + 32, 72, t * 16, lane);
      s[t] = wmma16x16x32(qa0, b0, vzero);
      s[t] = wmma16x16x32(qa1, b1, s[t]);
    }

    // scale + mask
#pragma unroll
    for (int t = 0; t < 4; ++t)
#pragma unroll
      for (int r = 0; r < 8; ++r) {
        int lrow = waveRow + r + 8 * hl;
        int col = cb + t * 16 + ln;
        bool ok = (col >= rp_lo[lrow] && col <= rp_hi[lrow]) ||
                  (col == rp_e1[lrow]) || (col == rp_e2[lrow]);
        s[t][r] = ok ? s[t][r] * 0.125f : -1e30f;
      }

    // online softmax (reductions over 16-lane halves; row = r + 8*hl)
    float newm[8], alpha[8];
#pragma unroll
    for (int r = 0; r < 8; ++r) {
      float m = fmaxf(fmaxf(s[0][r], s[1][r]), fmaxf(s[2][r], s[3][r]));
#pragma unroll
      for (int off = 1; off < 16; off <<= 1)
        m = fmaxf(m, __shfl_xor(m, off, 32));
      newm[r] = fmaxf(m_run[r], m);
      alpha[r] = __expf(m_run[r] - newm[r]);
      m_run[r] = newm[r];
    }
    float rsum[8];
#pragma unroll
    for (int r = 0; r < 8; ++r) rsum[r] = 0.0f;
#pragma unroll
    for (int t = 0; t < 4; ++t)
#pragma unroll
      for (int r = 0; r < 8; ++r) {
        float sv = s[t][r];
        float p = (sv <= -1e29f) ? 0.0f : __expf(sv - newm[r]);
        s[t][r] = p;
        rsum[r] += p;
      }
#pragma unroll
    for (int r = 0; r < 8; ++r) {
#pragma unroll
      for (int off = 1; off < 16; off <<= 1)
        rsum[r] += __shfl_xor(rsum[r], off, 32);
      l_run[r] = l_run[r] * alpha[r] + rsum[r];
    }
#pragma unroll
    for (int j = 0; j < 4; ++j)
#pragma unroll
      for (int r = 0; r < 8; ++r) o_acc[j][r] *= alpha[r];

    // write P to per-wave LDS slab, then O += P V
    _Float16* pw = Ps + wid * 16 * 72;
#pragma unroll
    for (int t = 0; t < 4; ++t)
#pragma unroll
      for (int r = 0; r < 8; ++r)
        pw[(r + 8 * hl) * 72 + t * 16 + ln] = (_Float16)s[t][r];
    __syncthreads();

    v16h pa0 = lds_afrag(pw, 72, 0, lane);
    v16h pa1 = lds_afrag(pw + 32, 72, 0, lane);
#pragma unroll
    for (int j = 0; j < 4; ++j) {
      v16h vb0 = lds_bfrag(Vts, 72, j * 16, lane);
      v16h vb1 = lds_bfrag(Vts + 32, 72, j * 16, lane);
      o_acc[j] = wmma16x16x32(pa0, vb0, o_acc[j]);
      o_acc[j] = wmma16x16x32(pa1, vb1, o_acc[j]);
    }
  }

#pragma unroll
  for (int j = 0; j < 4; ++j)
#pragma unroll
    for (int r = 0; r < 8; ++r) {
      int row = rowBlk + waveRow + r + 8 * hl;
      if (row < NTOK) {
        float v = o_acc[j][r] / l_run[r];
        attn16[(size_t)row * DIM + head * HEAD_DIM + j * 16 + ln] =
            (_Float16)v;
      }
    }
}

// ---------------------------------------------------------------- launcher
extern "C" void kernel_launch(void* const* d_in, const int* in_sizes, int n_in,
                              void* d_out, int out_size, void* d_ws,
                              size_t ws_size, hipStream_t stream) {
  const float* x = (const float*)d_in[0];
  const int* patch_idx = (const int*)d_in[1];
  const int* point_idx = (const int*)d_in[2];
  const float* ln1_w = (const float*)d_in[3];
  const float* ln1_b = (const float*)d_in[4];
  const float* qkv_w = (const float*)d_in[5];
  const float* proj_w = (const float*)d_in[6];
  const float* proj_b = (const float*)d_in[7];
  const float* ln2_w = (const float*)d_in[8];
  const float* ln2_b = (const float*)d_in[9];
  const float* fc1_w = (const float*)d_in[10];
  const float* fc1_b = (const float*)d_in[11];
  const float* fc2_w = (const float*)d_in[12];
  const float* fc2_b = (const float*)d_in[13];
  float* out = (float*)d_out;
  (void)in_sizes; (void)n_in; (void)out_size; (void)ws_size;

  char* ws = (char*)d_ws;
  size_t off = 0;
  auto carve = [&](size_t bytes) {
    void* p = ws + off;
    off = (off + bytes + 255) & ~(size_t)255;
    return p;
  };
  _Float16* qkvw16 = (_Float16*)carve((size_t)3 * DIM * DIM * 2);
  _Float16* projw16 = (_Float16*)carve((size_t)DIM * DIM * 2);
  _Float16* fc1w16 = (_Float16*)carve((size_t)HIDDEN * DIM * 2);
  _Float16* fc2w16 = (_Float16*)carve((size_t)DIM * HIDDEN * 2);
  _Float16* xn16 = (_Float16*)carve((size_t)NTOK * DIM * 2);
  _Float16* q16p = (_Float16*)carve((size_t)HEADS * NTOK * HEAD_DIM * 2);
  _Float16* k16p = (_Float16*)carve((size_t)HEADS * NTOK * HEAD_DIM * 2);
  _Float16* v16p = (_Float16*)carve((size_t)HEADS * NTOK * HEAD_DIM * 2);
  _Float16* attn16 = (_Float16*)carve((size_t)NTOK * DIM * 2);
  float* x1 = (float*)carve((size_t)NTOK * DIM * 4);
  _Float16* h16 = (_Float16*)carve((size_t)NTOK * DIM * 2);
  _Float16* h1_16 = (_Float16*)carve((size_t)NTOK * HIDDEN * 2);
  int* lohi = (int*)carve(2 * NPATCH * sizeof(int));

  f32_to_f16_kernel<<<(3 * DIM * DIM + 255) / 256, 256, 0, stream>>>(
      qkv_w, qkvw16, 3 * DIM * DIM);
  f32_to_f16_kernel<<<(DIM * DIM + 255) / 256, 256, 0, stream>>>(
      proj_w, projw16, DIM * DIM);
  f32_to_f16_kernel<<<(HIDDEN * DIM + 255) / 256, 256, 0, stream>>>(
      fc1_w, fc1w16, HIDDEN * DIM);
  f32_to_f16_kernel<<<(DIM * HIDDEN + 255) / 256, 256, 0, stream>>>(
      fc2_w, fc2w16, DIM * HIDDEN);
  seg_minmax_kernel<<<1, 256, 0, stream>>>(patch_idx, lohi);

  layernorm_f16_kernel<<<NTOK, 256, 0, stream>>>(x, ln1_w, ln1_b, xn16);

  dim3 gQKV((NTOK + 127) / 128, (3 * DIM) / 128);
  gemm_wmma_kernel<EPI_QKV><<<gQKV, 256, 0, stream>>>(
      xn16, qkvw16, NTOK, DIM, 3 * DIM, nullptr, nullptr, nullptr, nullptr,
      q16p, k16p, v16p);

  dim3 gAttn((NTOK + 127) / 128, HEADS);
  attn_flash_kernel<<<gAttn, 256, 0, stream>>>(q16p, k16p, v16p, point_idx,
                                               lohi, attn16);

  dim3 gProj((NTOK + 127) / 128, DIM / 128);
  gemm_wmma_kernel<EPI_PROJ><<<gProj, 256, 0, stream>>>(
      attn16, projw16, NTOK, DIM, DIM, proj_b, x, x1, nullptr, nullptr,
      nullptr, nullptr);

  layernorm_f16_kernel<<<NTOK, 256, 0, stream>>>(x1, ln2_w, ln2_b, h16);

  dim3 gFc1((NTOK + 127) / 128, HIDDEN / 128);
  gemm_wmma_kernel<EPI_FC1><<<gFc1, 256, 0, stream>>>(
      h16, fc1w16, NTOK, DIM, HIDDEN, fc1_b, nullptr, nullptr, h1_16, nullptr,
      nullptr, nullptr);

  dim3 gFc2((NTOK + 127) / 128, DIM / 128);
  gemm_wmma_kernel<EPI_FC2><<<gFc2, 256, 0, stream>>>(
      h1_16, fc2w16, NTOK, HIDDEN, DIM, fc2_b, x1, out, nullptr, nullptr,
      nullptr, nullptr);
}
